// Decoder_2130303779224
// MI455X (gfx1250) — matrix operand order repaired and checked by reading, not compiled, not run on hardware
//
#include <hip/hip_runtime.h>
#include <hip/hip_bf16.h>
#include <math.h>

#define B_   64
#define P_   196
#define ENC_ 512
#define EMB_ 256
#define ATT_ 256
#define DEC_ 256
#define V_   10000
#define T_   44

typedef __attribute__((ext_vector_type(16))) __bf16       v16bf;
typedef __attribute__((ext_vector_type(8)))  float        v8f;
typedef __attribute__((ext_vector_type(4)))  unsigned int u32x4;

union FragBF { v16bf v; u32x4 q[2]; };

// Async global->LDS staging if this toolchain exposes the gfx1250 builtins.
#if defined(__has_builtin)
#  if __has_builtin(__builtin_amdgcn_global_load_async_to_lds_b128) && \
      __has_builtin(__builtin_amdgcn_s_wait_asynccnt)
#    define USE_ASYNC_LDS 1
#  endif
#endif
#ifndef USE_ASYNC_LDS
#  define USE_ASYNC_LDS 0
#endif

#if USE_ASYNC_LDS
// Builtin expects pointers to 128-bit int vectors (per compiler diagnostic):
//   param0: global (AS1) v4i*, param1: LDS (AS3) v4i*
typedef int v4i_ __attribute__((vector_size(16)));
typedef __attribute__((address_space(1))) v4i_* as1_v4i;
typedef __attribute__((address_space(3))) v4i_* as3_v4i;
#endif

// ---------------------------------------------------------------------------
// Generic bf16 WMMA GEMM:  C[M,N] = A[M,K] * Bw[N,K]^T (+ bias[n])
// One wave computes one 16x16 tile; K stepped by 32 via v_wmma_f32_16x16x32_bf16.
// Used for the large-M att1 GEMM where operands are unique per tile.
// ---------------------------------------------------------------------------
__global__ void gemm_bf16_wmma(const __bf16* __restrict__ A, int lda,
                               const __bf16* __restrict__ Bw, int ldb,
                               const float* __restrict__ bias,
                               float* __restrict__ C, long ldc,
                               int mt, int nt, int K)
{
    const int lane = threadIdx.x & 31;
    const int wid  = threadIdx.x >> 5;
    const int tile = blockIdx.x * (blockDim.x >> 5) + wid;
    if (tile >= mt * nt) return;           // wave-uniform exit: EXEC stays all-1s
    const int tm = tile / nt;
    const int tn = tile % nt;
    const int r  = lane & 15;              // m-in-tile for A, n-in-tile for B/C
    const int g  = lane >> 4;              // half-wave group

    const __bf16* Ap = A  + (long)(tm * 16 + r) * lda;
    const __bf16* Bp = Bw + (long)(tn * 16 + r) * ldb;

    v8f acc = {};
    for (int k0 = 0; k0 < K; k0 += 32) {
        FragBF a, b;
        // A 16x32 bf16 layout: g=0 -> K 0..7 (V0..3), 16..23 (V4..7); g=1 -> +8
        a.q[0] = *(const u32x4*)(Ap + k0 + g * 8);
        a.q[1] = *(const u32x4*)(Ap + k0 + 16 + g * 8);
        // B 32x16 bf16 layout: same as A: g=0 -> K 0..7, 16..23; g=1 -> +8
        b.q[0] = *(const u32x4*)(Bp + k0 + g * 8);
        b.q[1] = *(const u32x4*)(Bp + k0 + 16 + g * 8);
        acc = __builtin_amdgcn_wmma_f32_16x16x32_bf16(false, a.v, false, b.v,
                                                      (short)0, acc, false, false);
    }
    const long  col = tn * 16 + r;
    const float bs  = bias ? bias[col] : 0.0f;
#pragma unroll
    for (int i = 0; i < 8; ++i) {
        const long row = tm * 16 + g * 8 + i;   // C/D: VGPR i -> M = i + g*8
        C[row * ldc + col] = acc[i] + bs;
    }
}

// ---------------------------------------------------------------------------
// M=64 WMMA GEMM with LDS-staged weight tile:
//   block = 4 waves = all 4 M-tiles; one 16-wide N-tile per block.
//   The 16xK weight slab is staged into LDS once (async if available) and
//   shared by the 4 waves -> 4x less L2 traffic on the dominant GEMM.
//   Rows padded by 8 bf16 so ds_load_b128 B-fragment reads are conflict-free.
// ---------------------------------------------------------------------------
extern __shared__ char smem_raw[];

__global__ void gemm_m64_bf16_wmma(const __bf16* __restrict__ A, int lda,
                                   const __bf16* __restrict__ Bw, int ldb,
                                   const float* __restrict__ bias,
                                   float* __restrict__ C, long ldc, int K)
{
    __bf16* Bs = (__bf16*)smem_raw;
    const int Kp   = K + 8;                // padded row stride (bf16 elems)
    const int tid  = threadIdx.x;
    const int lane = tid & 31;
    const int w    = tid >> 5;             // wave index = M-tile (0..3)
    const int r    = lane & 15;
    const int g    = lane >> 4;
    const int n0   = blockIdx.x * 16;

    // ---- stage B tile [16 x K] into LDS ----
    const int kq     = K >> 3;             // 16B chunks per row
    const int chunks = 16 * kq;
    for (int cidx = tid; cidx < chunks; cidx += 128) {
        int row = cidx / kq;
        int kc  = (cidx - row * kq) << 3;
#if USE_ASYNC_LDS
        __builtin_amdgcn_global_load_async_to_lds_b128(
            (as1_v4i)(Bw + (long)(n0 + row) * ldb + kc),
            (as3_v4i)(Bs + row * Kp + kc), 0, 0);
#else
        *(u32x4*)(Bs + row * Kp + kc) = *(const u32x4*)(Bw + (long)(n0 + row) * ldb + kc);
#endif
    }
#if USE_ASYNC_LDS
    __builtin_amdgcn_s_wait_asynccnt(0);
#endif
    __syncthreads();

    const __bf16* Ap = A  + (long)(w * 16 + r) * lda;   // activations: tiny, L2-hot
    const __bf16* Bp = Bs + r * Kp;                     // weight rows from LDS

    v8f acc = {};
    for (int k0 = 0; k0 < K; k0 += 32) {
        FragBF a, b;
        a.q[0] = *(const u32x4*)(Ap + k0 + g * 8);
        a.q[1] = *(const u32x4*)(Ap + k0 + 16 + g * 8);
        b.q[0] = *(const u32x4*)(Bp + k0 + g * 8);
        b.q[1] = *(const u32x4*)(Bp + k0 + 16 + g * 8);
        acc = __builtin_amdgcn_wmma_f32_16x16x32_bf16(false, a.v, false, b.v,
                                                      (short)0, acc, false, false);
    }
    const long  col = n0 + r;
    const float bs  = bias ? bias[col] : 0.0f;
#pragma unroll
    for (int i = 0; i < 8; ++i) {
        const long row = w * 16 + g * 8 + i;
        C[row * ldc + col] = acc[i] + bs;
    }
}

// ---------------------------------------------------------------------------
// fp32 -> bf16 conversion
// ---------------------------------------------------------------------------
__global__ void f32_to_bf16(const float* __restrict__ in, __bf16* __restrict__ out, long n)
{
    long i = (long)blockIdx.x * blockDim.x + threadIdx.x;
    if (i < n) out[i] = (__bf16)in[i];
}

// ---------------------------------------------------------------------------
// Fuse W_ih[1024,768] and W_hh[1024,256] into Wg[1024,1024] bf16, bias = b_ih+b_hh
// ---------------------------------------------------------------------------
__global__ void build_gates_w(const float* __restrict__ W_ih, const float* __restrict__ W_hh,
                              const float* __restrict__ b_ih, const float* __restrict__ b_hh,
                              __bf16* __restrict__ Wg, float* __restrict__ bg)
{
    int n = blockIdx.x;           // gate row, 0..1023
    int t = threadIdx.x;          // 0..255
    for (int k = t; k < ENC_ + EMB_; k += 256)
        Wg[(long)n * 1024 + k] = (__bf16)W_ih[(long)n * (ENC_ + EMB_) + k];
    Wg[(long)n * 1024 + (ENC_ + EMB_) + t] = (__bf16)W_hh[(long)n * DEC_ + t];
    if (t == 0) bg[n] = b_ih[n] + b_hh[n];
}

// ---------------------------------------------------------------------------
// h0/c0 from mean-pooled encoder; also seed the h-slot of xh (bf16)
// ---------------------------------------------------------------------------
__global__ void init_state(const float* __restrict__ enc,
                           const float* __restrict__ W_init_h, const float* __restrict__ b_init_h,
                           const float* __restrict__ W_init_c, const float* __restrict__ b_init_c,
                           float* __restrict__ h, float* __restrict__ c,
                           __bf16* __restrict__ xh)
{
    __shared__ float avg[ENC_];
    int b = blockIdx.x, t = threadIdx.x;
    for (int e = t; e < ENC_; e += 256) {
        float s = 0.0f;
        const float* p = enc + ((long)b * P_) * ENC_ + e;
        for (int j = 0; j < P_; ++j) s += p[(long)j * ENC_];
        avg[e] = s * (1.0f / (float)P_);
    }
    __syncthreads();
    float sh = b_init_h[t], sc = b_init_c[t];
    const float* wh = W_init_h + (long)t * ENC_;
    const float* wc = W_init_c + (long)t * ENC_;
    for (int e = 0; e < ENC_; ++e) { sh += avg[e] * wh[e]; sc += avg[e] * wc[e]; }
    h[b * DEC_ + t] = sh;
    c[b * DEC_ + t] = sc;
    xh[(long)b * 1024 + (ENC_ + EMB_) + t] = (__bf16)sh;
}

// ---------------------------------------------------------------------------
// Per-step attention: att2 GEMV, relu scores, softmax over P, context,
// write alphas to output and [ctx | emb_t] (bf16) into xh.
// ---------------------------------------------------------------------------
__global__ void attention_step(const float* __restrict__ h, const float* __restrict__ att1,
                               const float* __restrict__ W_dec_att, const float* __restrict__ W_full,
                               const float* __restrict__ enc, const float* __restrict__ embedding,
                               const int* __restrict__ caption,
                               __bf16* __restrict__ xh, float* __restrict__ alphas, int t)
{
    __shared__ float att2s[ATT_];
    __shared__ float sc[256];
    __shared__ float red[256];
    int b = blockIdx.x, tid = threadIdx.x;

    // att2[a] = h[b] . W_dec_att[a]
    {
        const float* hb = h + (long)b * DEC_;
        const float* w  = W_dec_att + (long)tid * DEC_;
        float s = 0.0f;
        for (int d = 0; d < DEC_; ++d) s += hb[d] * w[d];
        att2s[tid] = s;
    }
    __syncthreads();

    // scores[p] = sum_a relu(att1 + att2) * W_full
    float sv = -INFINITY;
    if (tid < P_) {
        const float* a1 = att1 + ((long)b * P_ + tid) * ATT_;
        float s = 0.0f;
        for (int a = 0; a < ATT_; ++a) {
            float v = a1[a] + att2s[a];
            s += (v > 0.0f ? v : 0.0f) * W_full[a];
        }
        sv = s;
    }
    red[tid] = sv;
    __syncthreads();
    for (int o = 128; o > 0; o >>= 1) {
        if (tid < o) red[tid] = fmaxf(red[tid], red[tid + o]);
        __syncthreads();
    }
    float mx = red[0];
    __syncthreads();
    float ev = (tid < P_) ? expf(sv - mx) : 0.0f;
    red[tid] = ev;
    __syncthreads();
    for (int o = 128; o > 0; o >>= 1) {
        if (tid < o) red[tid] += red[tid + o];
        __syncthreads();
    }
    float inv = 1.0f / red[0];
    __syncthreads();
    if (tid < P_) {
        float a = ev * inv;
        sc[tid] = a;
        alphas[((long)b * T_ + t) * P_ + tid] = a;
    }
    __syncthreads();

    // ctx[e] = sum_p alpha[p] * enc[b,p,e]  -> xh[0..511] (bf16)
    for (int e = tid; e < ENC_; e += 256) {
        const float* pe = enc + ((long)b * P_) * ENC_ + e;
        float s = 0.0f;
        for (int p = 0; p < P_; ++p) s += sc[p] * pe[(long)p * ENC_];
        xh[(long)b * 1024 + e] = (__bf16)s;
    }
    // embedding lookup -> xh[512..767]
    int cap = caption[b * T_ + t];
    xh[(long)b * 1024 + ENC_ + tid] = (__bf16)embedding[(long)cap * EMB_ + tid];
}

// ---------------------------------------------------------------------------
// LSTM cell nonlinearity (torch gate order i,f,g,o); writes h,c and bf16 copies
// ---------------------------------------------------------------------------
__global__ void lstm_cell(const float* __restrict__ gates, float* __restrict__ c,
                          float* __restrict__ h, __bf16* __restrict__ xh,
                          __bf16* __restrict__ hnB)
{
    int i = blockIdx.x * blockDim.x + threadIdx.x;   // B*DEC
    int b = i >> 8, d = i & 255;
    const float* g = gates + (long)b * (4 * DEC_);
    float ig = 1.0f / (1.0f + expf(-g[d]));
    float fg = 1.0f / (1.0f + expf(-g[DEC_ + d]));
    float gg = tanhf(g[2 * DEC_ + d]);
    float og = 1.0f / (1.0f + expf(-g[3 * DEC_ + d]));
    float cn = fg * c[i] + ig * gg;
    float hn = og * tanhf(cn);
    c[i] = cn;
    h[i] = hn;
    xh[(long)b * 1024 + (ENC_ + EMB_) + d] = (__bf16)hn;
    hnB[i] = (__bf16)hn;
}

// ---------------------------------------------------------------------------
extern "C" void kernel_launch(void* const* d_in, const int* in_sizes, int n_in,
                              void* d_out, int out_size, void* d_ws, size_t ws_size,
                              hipStream_t stream)
{
    const float* enc       = (const float*)d_in[0];
    const int*   caption   = (const int*)  d_in[1];
    const float* W_enc_att = (const float*)d_in[2];
    const float* W_dec_att = (const float*)d_in[3];
    const float* W_full    = (const float*)d_in[4];
    const float* embedding = (const float*)d_in[5];
    const float* W_init_h  = (const float*)d_in[6];
    const float* b_init_h  = (const float*)d_in[7];
    const float* W_init_c  = (const float*)d_in[8];
    const float* b_init_c  = (const float*)d_in[9];
    const float* W_ih      = (const float*)d_in[10];
    const float* b_ih      = (const float*)d_in[11];
    const float* W_hh      = (const float*)d_in[12];
    const float* b_hh      = (const float*)d_in[13];
    const float* W_out     = (const float*)d_in[14];
    const float* b_out     = (const float*)d_in[15];

    float* pred = (float*)d_out;                       // [B,T,V]
    float* alph = pred + (size_t)B_ * T_ * V_;         // [B,T,P]

    char*  ws  = (char*)d_ws;
    size_t off = 0;
    auto carve = [&](size_t bytes) -> void* {
        off = (off + 255) & ~(size_t)255;
        void* p = ws + off;
        off += bytes;
        return p;
    };

    __bf16* encB   = (__bf16*)carve((size_t)B_ * P_ * ENC_ * 2);
    __bf16* WencB  = (__bf16*)carve((size_t)ATT_ * ENC_ * 2);
    __bf16* WgB    = (__bf16*)carve((size_t)1024 * 1024 * 2);
    __bf16* WoutB  = (__bf16*)carve((size_t)V_ * DEC_ * 2);
    __bf16* xh     = (__bf16*)carve((size_t)B_ * 1024 * 2);
    __bf16* hnB    = (__bf16*)carve((size_t)B_ * DEC_ * 2);
    float*  att1   = (float*) carve((size_t)B_ * P_ * ATT_ * 4);
    float*  h      = (float*) carve((size_t)B_ * DEC_ * 4);
    float*  c      = (float*) carve((size_t)B_ * DEC_ * 4);
    float*  gates  = (float*) carve((size_t)B_ * 4 * DEC_ * 4);
    float*  bgates = (float*) carve((size_t)4 * DEC_ * 4);

    // ---- one-time prep ----
    long nEnc = (long)B_ * P_ * ENC_;
    f32_to_bf16<<<(unsigned)((nEnc + 255) / 256), 256, 0, stream>>>(enc, encB, nEnc);
    f32_to_bf16<<<(ATT_ * ENC_ + 255) / 256, 256, 0, stream>>>(W_enc_att, WencB, (long)ATT_ * ENC_);
    long nWo = (long)V_ * DEC_;
    f32_to_bf16<<<(unsigned)((nWo + 255) / 256), 256, 0, stream>>>(W_out, WoutB, nWo);
    build_gates_w<<<1024, 256, 0, stream>>>(W_ih, W_hh, b_ih, b_hh, WgB, bgates);
    init_state<<<B_, 256, 0, stream>>>(enc, W_init_h, b_init_h, W_init_c, b_init_c, h, c, xh);

    // att1 = enc @ W_enc_att^T   (M = B*P = 12544, N = 256, K = 512)
    {
        int mt = (B_ * P_) / 16, nt = ATT_ / 16, tiles = mt * nt;
        gemm_bf16_wmma<<<(tiles + 3) / 4, 128, 0, stream>>>(
            encB, ENC_, WencB, ENC_, nullptr, att1, ATT_, mt, nt, ENC_);
    }

    const size_t smemGates  = (size_t)16 * (1024 + 8) * sizeof(__bf16);
    const size_t smemLogits = (size_t)16 * (DEC_ + 8) * sizeof(__bf16);

    // ---- sequential decode ----
    for (int t = 0; t < T_; ++t) {
        attention_step<<<B_, 256, 0, stream>>>(h, att1, W_dec_att, W_full, enc,
                                               embedding, caption, xh, alph, t);
        // gates = xh[64,1024] @ WgB[1024,1024]^T + (b_ih + b_hh)
        gemm_m64_bf16_wmma<<<(4 * DEC_) / 16, 128, smemGates, stream>>>(
            xh, 1024, WgB, 1024, bgates, gates, 4 * DEC_, 1024);
        lstm_cell<<<(B_ * DEC_) / 256, 256, 0, stream>>>(gates, c, h, xh, hnB);
        // logits -> predictions[:, t, :]   (M = 64, N = 10000, K = 256)
        gemm_m64_bf16_wmma<<<V_ / 16, 128, smemLogits, stream>>>(
            hnB, DEC_, WoutB, DEC_, b_out, pred + (long)t * V_, (long)T_ * V_, DEC_);
    }
}
